// MultiHeadSelfAttention_1511828488284
// MI455X (gfx1250) — matrix operand order repaired and checked by reading, not compiled, not run on hardware
//
#include <hip/hip_runtime.h>
#include <stdint.h>

// MI455X (gfx1250) multi-head self-attention, flash-attention style.
// B=8, S=2048, H=256, NH=8, HD=32.
// Matmuls: v_wmma_f32_16x16x32_f16. K/V staged via global_load_async_to_lds_b128
// (ASYNCcnt) into double-buffered LDS with copy/compute overlap.

typedef _Float16 half_t;
typedef __attribute__((ext_vector_type(16))) _Float16 v16h;
typedef __attribute__((ext_vector_type(8)))  _Float16 v8h;
typedef __attribute__((ext_vector_type(8)))  float    v8f;

#define B_DIM  8
#define S_DIM  2048
#define H_DIM  256
#define NH_DIM 8
#define HD_DIM 32

static __device__ __forceinline__ v8f wmma_f16(v16h a, v16h b, v8f c) {
  // D = A(16x32 f16) * B(32x16 f16) + C(16x16 f32)
  return __builtin_amdgcn_wmma_f32_16x16x32_f16(false, a, false, b, (short)0, c,
                                                false, false);
}

// Async copy 16 bytes global -> LDS (tracked by ASYNCcnt).
// lds_byte = wave-relative LDS byte offset (low 32 bits of generic shared addr).
static __device__ __forceinline__ void async_copy_b128(uint32_t lds_byte,
                                                       const void* gaddr) {
  asm volatile("global_load_async_to_lds_b128 %0, %1, off"
               :
               : "v"(lds_byte), "v"(gaddr)
               : "memory");
}

static __device__ __forceinline__ void wait_async0() {
#if __has_builtin(__builtin_amdgcn_s_wait_asynccnt)
  __builtin_amdgcn_s_wait_asynccnt(0);
#else
  asm volatile("s_wait_asynccnt 0x0" ::: "memory");
#endif
}

// A fragment (16x32, f16, row-major source, row stride = ld elements).
// Lane L: m = L&15, h = L>>4. elems 0..7 -> k = 8h..8h+7 ; elems 8..15 -> 16+8h..16+8h+7
static __device__ __forceinline__ v16h load_frag_a(const half_t* __restrict__ base, int ld) {
  const int lane = threadIdx.x & 31;
  const int m  = lane & 15;
  const int hh = lane >> 4;
  const half_t* p = base + m * ld + 8 * hh;
  v8h lo = *(const v8h*)(p);
  v8h hi = *(const v8h*)(p + 16);
  v16h r;
#pragma unroll
  for (int i = 0; i < 8; ++i) { r[i] = lo[i]; r[i + 8] = hi[i]; }
  return r;
}

// B fragment (32x16, f16) from n-major storage: src[n][k], row stride = ld elements.
// Lane L: n = L&15, h = L>>4. elems 0..7 -> k = 8h..8h+7 ; elems 8..15 -> 16+8h..16+8h+7
static __device__ __forceinline__ v16h load_frag_b(const half_t* __restrict__ base, int ld) {
  const int lane = threadIdx.x & 31;
  const int n  = lane & 15;
  const int hh = lane >> 4;
  const half_t* p = base + n * ld + 8 * hh;
  v8h lo = *(const v8h*)(p);
  v8h hi = *(const v8h*)(p + 16);
  v16h r;
#pragma unroll
  for (int i = 0; i < 8; ++i) { r[i] = lo[i]; r[i + 8] = hi[i]; }
  return r;
}

// ---------------------------------------------------------------------------
// Pre-pass: f32 -> f16 cast (contiguous)
__global__ void __launch_bounds__(256) cast_f16_kernel(
    const float* __restrict__ in, half_t* __restrict__ out, int n) {
  int i = blockIdx.x * 256 + threadIdx.x;
  if (i < n) out[i] = (half_t)in[i];
}

// Pre-pass: transpose-cast w[K][N] f32 -> wt[N][K] f16 (n-major for B fragments)
__global__ void __launch_bounds__(256) transpose_cast_kernel(
    const float* __restrict__ in, half_t* __restrict__ out, int K, int N) {
  int i = blockIdx.x * 256 + threadIdx.x;
  if (i < K * N) {
    int k = i / N;
    int n = i - k * N;
    out[n * K + k] = (half_t)in[i];
  }
}

// ---------------------------------------------------------------------------
// Wave computes a 16x64 output tile: C = A(16xK) * Bt(64xK)^T, K = KDIM.
template <int KDIM>
static __device__ __forceinline__ void wave_gemm_16x64(
    const half_t* __restrict__ A, const half_t* __restrict__ Bt, v8f acc[4]) {
#pragma unroll
  for (int k = 0; k < KDIM; k += 32) {
    v16h a = load_frag_a(A + k, KDIM);
#pragma unroll
    for (int j = 0; j < 4; ++j) {
      v16h b = load_frag_b(Bt + (j * 16) * KDIM + k, KDIM);
      acc[j] = wmma_f16(a, b, acc[j]);
    }
  }
}

// ---------------------------------------------------------------------------
// Kernel 1: QKV projection. x_h[16384,256] @ w_qkv_t[768,256]^T + b_qkv.
// Scatter: Q,K -> [b][nh][s][hd] f16 ; V -> [b][nh][hd][s] f16 (transposed).
__global__ void __launch_bounds__(256) qkv_gemm_kernel(
    const half_t* __restrict__ xh, const half_t* __restrict__ wqt,
    const float* __restrict__ bias,
    half_t* __restrict__ q_ws, half_t* __restrict__ k_ws, half_t* __restrict__ v_ws) {
  const int wave = threadIdx.x >> 5;
  const int lane = threadIdx.x & 31;
  const int nloc = lane & 15;
  const int hh   = lane >> 4;
  const int m0 = (blockIdx.x * 8 + wave) * 16;
  const int n0 = blockIdx.y * 64;

  v8f acc[4];
#pragma unroll
  for (int j = 0; j < 4; ++j)
#pragma unroll
    for (int r = 0; r < 8; ++r) acc[j][r] = 0.0f;

  wave_gemm_16x64<H_DIM>(xh + (size_t)m0 * H_DIM, wqt + (size_t)n0 * H_DIM, acc);

#pragma unroll
  for (int j = 0; j < 4; ++j) {
    const int n      = n0 + j * 16 + nloc;
    const float bs   = bias[n];
    const int region = n >> 8;          // 0=Q 1=K 2=V
    const int nh     = (n & 255) >> 5;
    const int hd     = n & 31;
#pragma unroll
    for (int r = 0; r < 8; ++r) {
      const int row = m0 + hh * 8 + r;
      const int b   = row >> 11;        // / S_DIM
      const int s   = row & (S_DIM - 1);
      const half_t val = (half_t)(acc[j][r] + bs);
      if (region == 0)
        q_ws[(((size_t)(b * NH_DIM + nh)) * S_DIM + s) * HD_DIM + hd] = val;
      else if (region == 1)
        k_ws[(((size_t)(b * NH_DIM + nh)) * S_DIM + s) * HD_DIM + hd] = val;
      else
        v_ws[(((size_t)(b * NH_DIM + nh)) * HD_DIM + hd) * S_DIM + s] = val;
    }
  }
}

// ---------------------------------------------------------------------------
// Kernel 2: flash attention. 4 waves/block, each wave = 16 query rows.
// grid = B*NH*(S/64). K/V tiles async-staged into double-buffered LDS and
// shared by all 4 waves. Output ctx -> [b][s][nh][hd] f16 (== [b][s][h]).
__global__ void __launch_bounds__(128) attn_kernel(
    const half_t* __restrict__ q_ws, const half_t* __restrict__ k_ws,
    const half_t* __restrict__ v_ws, half_t* __restrict__ ctx_ws) {
  __shared__ half_t kbuf[2][64 * HD_DIM];   // [key][hd], ld=32 ; 4KB each
  __shared__ half_t vbuf[2][HD_DIM * 64];   // [hd][key], ld=64 ; 4KB each
  __shared__ half_t pbuf[4][16 * 64];       // wave-private P staging

  const int tid  = threadIdx.x;
  const int wave = tid >> 5;
  const int lane = tid & 31;
  const int nloc = lane & 15;
  const int hh   = lane >> 4;

  const int t  = blockIdx.x;
  const int qt = t & 31;                // S/64 = 32 tiles
  const int nh = (t >> 5) & 7;
  const int b  = t >> 8;
  const int q0 = qt * 64 + wave * 16;

  const half_t* __restrict__ Q  = q_ws + (((size_t)(b * NH_DIM + nh)) * S_DIM + q0) * HD_DIM;
  const half_t* __restrict__ K  = k_ws + ((size_t)(b * NH_DIM + nh)) * S_DIM * HD_DIM;
  const half_t* __restrict__ Vt = v_ws + ((size_t)(b * NH_DIM + nh)) * HD_DIM * S_DIM;
  half_t* __restrict__ P = &pbuf[wave][0];

  const uint32_t lds_k0 = (uint32_t)(uintptr_t)(&kbuf[0][0]);
  const uint32_t lds_k1 = (uint32_t)(uintptr_t)(&kbuf[1][0]);
  const uint32_t lds_v0 = (uint32_t)(uintptr_t)(&vbuf[0][0]);
  const uint32_t lds_v1 = (uint32_t)(uintptr_t)(&vbuf[1][0]);

  // Cooperative async stage of one 64-key tile (K: 4KB contiguous, V: 32 rows x 128B).
  auto issue_tile = [&](int kt, int bufi) {
    const int key0 = kt * 64;
    const uint32_t lk = bufi ? lds_k1 : lds_k0;
    const uint32_t lv = bufi ? lds_v1 : lds_v0;
    // K tile: 4096 bytes contiguous -> 256 x b128, 2 per thread
    {
      const char* gk = (const char*)(K + (size_t)key0 * HD_DIM);
      const int o = tid * 32;
      async_copy_b128(lk + o,      gk + o);
      async_copy_b128(lk + o + 16, gk + o + 16);
    }
    // V tile: rows hd=0..31, 128 bytes per row (row stride S_DIM*2 bytes)
    {
#pragma unroll
      for (int u = 0; u < 2; ++u) {
        const int c   = tid * 2 + u;        // 0..255
        const int row = c >> 3;             // 0..31
        const int col = (c & 7) * 16;       // byte offset in row
        const char* gv = (const char*)(Vt + (size_t)row * S_DIM + key0) + col;
        async_copy_b128(lv + row * 128 + col, gv);
      }
    }
  };

  issue_tile(0, 0);

  const v16h qa = load_frag_a(Q, HD_DIM);   // held in registers for the whole key loop

  v8f ctx0, ctx1;
  float row_m[8], row_l[8];
#pragma unroll
  for (int r = 0; r < 8; ++r) {
    ctx0[r] = 0.0f; ctx1[r] = 0.0f;
    row_m[r] = -1e30f; row_l[r] = 0.0f;
  }
  const float scale = 0.17677669529663687f;  // 1/sqrt(32)

  const int NT = S_DIM / 64;
  for (int kt = 0; kt < NT; ++kt) {
    const int bufi = kt & 1;

    wait_async0();        // this wave's async copies of tile kt complete
    __syncthreads();      // all waves' copies complete; prev compute done

    if (kt + 1 < NT) issue_tile(kt + 1, bufi ^ 1);   // overlap copy with compute

    const half_t* kt_lds = &kbuf[bufi][0];
    const half_t* vt_lds = &vbuf[bufi][0];

    // ---- scores: 4 x (16q x 16k) tiles, one WMMA each (K-dim = HD = 32)
    v8f sc[4];
#pragma unroll
    for (int j = 0; j < 4; ++j) {
      v16h kb = load_frag_b(kt_lds + (j * 16) * HD_DIM, HD_DIM);
      v8f z;
#pragma unroll
      for (int r = 0; r < 8; ++r) z[r] = 0.0f;
      sc[j] = wmma_f16(qa, kb, z);
    }

    // ---- online softmax (row = query; a row lives across 16 lanes of a half-wave)
#pragma unroll
    for (int r = 0; r < 8; ++r) {
      float pm = -1e30f;
#pragma unroll
      for (int j = 0; j < 4; ++j) {
        sc[j][r] *= scale;
        pm = fmaxf(pm, sc[j][r]);
      }
#pragma unroll
      for (int off = 8; off > 0; off >>= 1) pm = fmaxf(pm, __shfl_xor(pm, off));
      const float mnew  = fmaxf(row_m[r], pm);
      const float alpha = __expf(row_m[r] - mnew);
      row_m[r] = mnew;

      float ps = 0.0f;
#pragma unroll
      for (int j = 0; j < 4; ++j) {
        const float p = __expf(sc[j][r] - mnew);
        ps += p;
        P[(hh * 8 + r) * 64 + j * 16 + nloc] = (half_t)p;  // C-layout scatter to LDS
      }
#pragma unroll
      for (int off = 8; off > 0; off >>= 1) ps += __shfl_xor(ps, off);
      row_l[r] = row_l[r] * alpha + ps;
      ctx0[r] *= alpha;
      ctx1[r] *= alpha;
    }

    // ---- ctx += P(16x64) @ V(64x32): 2 K-chunks x 2 hd-tiles = 4 WMMAs
#pragma unroll
    for (int kk = 0; kk < 2; ++kk) {
      v16h pa  = load_frag_a(P + kk * 32, 64);                 // from LDS (wave-private)
      v16h vb0 = load_frag_b(vt_lds + kk * 32, 64);            // hd 0..15
      v16h vb1 = load_frag_b(vt_lds + 16 * 64 + kk * 32, 64);  // hd 16..31
      ctx0 = wmma_f16(pa, vb0, ctx0);
      ctx1 = wmma_f16(pa, vb1, ctx1);
    }
  }

  // ---- finalize: divide by l, store [b][s][nh*32+hd]
#pragma unroll
  for (int r = 0; r < 8; ++r) {
    const float inv = 1.0f / row_l[r];
    const int qrow  = q0 + hh * 8 + r;
    const size_t base = ((size_t)b * S_DIM + qrow) * H_DIM + nh * HD_DIM;
    ctx_ws[base + nloc]      = (half_t)(ctx0[r] * inv);
    ctx_ws[base + 16 + nloc] = (half_t)(ctx1[r] * inv);
  }
}

// ---------------------------------------------------------------------------
// Kernel 3: output projection. ctx_h[16384,256] @ w_out_t[256,256]^T + b_out -> f32 out.
__global__ void __launch_bounds__(256) out_gemm_kernel(
    const half_t* __restrict__ ctxh, const half_t* __restrict__ wot,
    const float* __restrict__ bias, float* __restrict__ out) {
  const int wave = threadIdx.x >> 5;
  const int lane = threadIdx.x & 31;
  const int nloc = lane & 15;
  const int hh   = lane >> 4;
  const int m0 = (blockIdx.x * 8 + wave) * 16;
  const int n0 = blockIdx.y * 64;

  v8f acc[4];
#pragma unroll
  for (int j = 0; j < 4; ++j)
#pragma unroll
    for (int r = 0; r < 8; ++r) acc[j][r] = 0.0f;

  wave_gemm_16x64<H_DIM>(ctxh + (size_t)m0 * H_DIM, wot + (size_t)n0 * H_DIM, acc);

#pragma unroll
  for (int j = 0; j < 4; ++j) {
    const int n    = n0 + j * 16 + nloc;
    const float bs = bias[n];
#pragma unroll
    for (int r = 0; r < 8; ++r) {
      const int row = m0 + hh * 8 + r;
      out[(size_t)row * H_DIM + n] = acc[j][r] + bs;
    }
  }
}

// ---------------------------------------------------------------------------
extern "C" void kernel_launch(void* const* d_in, const int* in_sizes, int n_in,
                              void* d_out, int out_size, void* d_ws, size_t ws_size,
                              hipStream_t stream) {
  const float* x     = (const float*)d_in[0];   // [8,2048,256]
  const float* w_qkv = (const float*)d_in[1];   // [256,768]
  const float* b_qkv = (const float*)d_in[2];   // [768]
  const float* w_out = (const float*)d_in[3];   // [256,256]
  const float* b_out = (const float*)d_in[4];   // [256]
  float* out = (float*)d_out;                   // [8,2048,256]

  char* ws = (char*)d_ws;
  const size_t M = (size_t)B_DIM * S_DIM;               // 16384
  size_t off = 0;
  half_t* xh  = (half_t*)(ws + off); off += M * H_DIM * 2;                       // 8 MB
  half_t* wqt = (half_t*)(ws + off); off += (size_t)3 * H_DIM * H_DIM * 2;       // 384 KB
  half_t* wot = (half_t*)(ws + off); off += (size_t)H_DIM * H_DIM * 2;           // 128 KB
  half_t* qws = (half_t*)(ws + off); off += M * H_DIM * 2;                       // 8 MB
  half_t* kws = (half_t*)(ws + off); off += M * H_DIM * 2;                       // 8 MB
  half_t* vws = (half_t*)(ws + off); off += M * H_DIM * 2;                       // 8 MB
  half_t* ctxh = (half_t*)(ws + off); off += M * H_DIM * 2;                      // 8 MB
  (void)ws_size; (void)in_sizes; (void)n_in; (void)out_size;

  // Pre-pass casts
  {
    int n = (int)(M * H_DIM);
    cast_f16_kernel<<<(n + 255) / 256, 256, 0, stream>>>(x, xh, n);
  }
  transpose_cast_kernel<<<(H_DIM * 3 * H_DIM + 255) / 256, 256, 0, stream>>>(
      w_qkv, wqt, H_DIM, 3 * H_DIM);
  transpose_cast_kernel<<<(H_DIM * H_DIM + 255) / 256, 256, 0, stream>>>(
      w_out, wot, H_DIM, H_DIM);

  // QKV projection: grid (M/128, 768/64)
  qkv_gemm_kernel<<<dim3(128, 12), 256, 0, stream>>>(xh, wqt, b_qkv, qws, kws, vws);

  // Flash attention: B*NH*(S/64) = 2048 blocks, 4 waves each
  attn_kernel<<<B_DIM * NH_DIM * (S_DIM / 64), 128, 0, stream>>>(qws, kws, vws, ctxh);

  // Output projection: grid (M/128, 256/64)
  out_gemm_kernel<<<dim3(128, 4), 256, 0, stream>>>(ctxh, wot, b_out, out);
}
